// NewEncoderLSTM_75531294867664
// MI455X (gfx1250) — matrix operand order repaired and checked by reading, not compiled, not run on hardware
//
#include <hip/hip_runtime.h>
#include <hip/hip_bf16.h>
#include <math.h>

// ---------------------------------------------------------------------------
// Types for CDNA5 WMMA (wave32)
// ---------------------------------------------------------------------------
typedef __bf16 bf16_t;
typedef __attribute__((ext_vector_type(16))) __bf16 v16bf;
typedef __attribute__((ext_vector_type(8)))  __bf16 v8bf;
typedef __attribute__((ext_vector_type(8)))  float  v8f;

// ---------------------------------------------------------------------------
// Problem geometry
// ---------------------------------------------------------------------------
#define T_STEPS 200
#define BATCH   256
#define XD      24
#define AD      8
#define IND     32          // XD + AD
#define HID     512
#define G4      2048        // 4*HID
#define KDIM    544         // HID + IND
#define OUTD    64

// Per-workgroup tiling for the recurrent GEMM
#define BT      32          // batch rows per WG (2 WMMA M-tiles)
#define HU      64          // hidden units per WG (4 unit-tiles of 16)
#define NB      (BATCH/BT)  // 8 batch groups
#define NH      (HID/HU)    // 8 hidden groups
#define LCOLS   (4*HU)      // 256 local gate columns (i|f|g|o blocks of 64)
#define THREADS 256         // 8 wave32s

// ---------------------------------------------------------------------------
// Workspace layout (bytes)
// ---------------------------------------------------------------------------
#define CNT_OFF    0                         // 64 ints (phase counters)
#define H_OFF      4096                      // double-buffered bf16 h state
#define HBUF_BYTES (BATCH*HID*2)             // 262144
#define W_OFF      (H_OFF + 2*HBUF_BYTES)    // bf16 [Whh|Wih] col-major, K=544
#define W_BYTES    (G4*KDIM*2)               // 2228224
#define HW_OFF     (W_OFF + W_BYTES)         // bf16 head weights (lin then lv)
#define HW_BYTES   (2*OUTD*HID*2)            // 131072

// ---------------------------------------------------------------------------
// Kernel 0: build bf16 weight images + zero state/counters (every launch)
// ---------------------------------------------------------------------------
__global__ void prep_kernel(const float* __restrict__ W_ih,
                            const float* __restrict__ W_hh,
                            const float* __restrict__ lin_w,
                            const float* __restrict__ lv_w,
                            char* __restrict__ ws) {
  int*    cnt  = (int*)(ws + CNT_OFF);
  bf16_t* hbuf = (bf16_t*)(ws + H_OFF);
  bf16_t* Wbf  = (bf16_t*)(ws + W_OFF);
  bf16_t* HWbf = (bf16_t*)(ws + HW_OFF);
  int tid    = blockIdx.x * blockDim.x + threadIdx.x;
  int stride = gridDim.x * blockDim.x;

  if (tid < 64) cnt[tid] = 0;
  for (int i = tid; i < 2 * BATCH * HID; i += stride) hbuf[i] = (bf16_t)0.0f;

  // Column-major combined weights: Wbf[col*KDIM + k], k<HID -> W_hh, else W_ih
  for (int i = tid; i < G4 * KDIM; i += stride) {
    int col = i / KDIM;
    int k   = i - col * KDIM;
    float v = (k < HID) ? W_hh[col * HID + k] : W_ih[col * IND + (k - HID)];
    Wbf[i] = (bf16_t)v;
  }
  // Head weights: [OUT][HID] row-major == B-column-major with K contiguous
  for (int i = tid; i < OUTD * HID; i += stride) {
    HWbf[i]              = (bf16_t)lin_w[i];
    HWbf[OUTD * HID + i] = (bf16_t)lv_w[i];
  }
}

// ---------------------------------------------------------------------------
// Kernel 1: persistent LSTM over 200 steps.
// Grid = NB*NH = 64 WGs; WG (bt,hg) owns batch rows [bt*32,..) and hidden
// units [hg*64,..). The full 278KB bf16 weight slice lives in the 320KB WGP
// LDS; A (=[h|obs]) is re-staged each step; i/f/g/o accumulate in registers
// of the same lane so the cell update and cell state never leave VGPRs.
// The K-loop is software-pipelined one stage deep so B fragments for K+32
// are in flight while the four WMMAs for K execute.
// ---------------------------------------------------------------------------
__device__ __forceinline__ float sigmoidf_fast(float x) {
  return 1.0f / (1.0f + __expf(-x));
}

__launch_bounds__(THREADS, 1)
__global__ void lstm_kernel(const float* __restrict__ x,
                            const float* __restrict__ a,
                            const float* __restrict__ mask,
                            const float* __restrict__ b_ih,
                            const float* __restrict__ b_hh,
                            char* __restrict__ ws) {
  __shared__ bf16_t Ws[LCOLS * KDIM];  // 278528 B : weight slice, col-major
  __shared__ bf16_t As[BT * KDIM];     //  34816 B : A = [h | masked obs]
  __shared__ float  bsum[LCOLS];       //   1024 B : b_ih + b_hh slice

  const int bt   = blockIdx.x % NB;
  const int hg   = blockIdx.x / NB;
  const int tid  = threadIdx.x;
  const int lane = tid & 31;
  const int wave = tid >> 5;

  int*          cnt  = (int*)(ws + CNT_OFF);
  bf16_t*       hbuf = (bf16_t*)(ws + H_OFF);
  const bf16_t* Wbf  = (const bf16_t*)(ws + W_OFF);

  // ---- one-time: bias sums + weight slice into LDS (col-major, K contig) --
  for (int n = tid; n < LCOLS; n += THREADS) {
    int gcol = (n >> 6) * HID + hg * HU + (n & 63);   // gate=n>>6, unit=n&63
    bsum[n] = b_ih[gcol] + b_hh[gcol];
  }
  for (int n = wave; n < LCOLS; n += 8) {
    int gcol = (n >> 6) * HID + hg * HU + (n & 63);
    const bf16_t* src = Wbf + (size_t)gcol * KDIM;
    bf16_t*       dst = Ws + (size_t)n * KDIM;
    for (int ch = lane; ch < KDIM / 8; ch += 32)
      *(v8bf*)(dst + ch * 8) = *(const v8bf*)(src + ch * 8);
  }
  __syncthreads();

  // ---- wave-fixed WMMA addressing (ISA 7.12.2) ----
  const int mtile = wave >> 2;          // 0..1 : which 16 batch rows
  const int utile = wave & 3;           // 0..3 : which 16 hidden units
  const int row   = lane & 15;
  const int k0off = (lane >> 4) * 16;   // A frag: lanes>=16 hold K+16..31
  const int kh16  = (lane >> 4) * 16;   // B frag: lanes>=16 hold K+16..31
  const int nloc  = utile * 16 + row;   // unit column within [0,64)
  // LDS base offsets of the 4 gate columns (i,f,g,o) for this lane's column
  const bf16_t* wI = Ws + (size_t)(0 * HU + nloc) * KDIM + kh16;
  const bf16_t* wF = Ws + (size_t)(1 * HU + nloc) * KDIM + kh16;
  const bf16_t* wG = Ws + (size_t)(2 * HU + nloc) * KDIM + kh16;
  const bf16_t* wO = Ws + (size_t)(3 * HU + nloc) * KDIM + kh16;
  const bf16_t* arow = As + (size_t)(mtile * 16 + row) * KDIM + k0off;

  const float bI = bsum[0 * HU + nloc];
  const float bF = bsum[1 * HU + nloc];
  const float bG = bsum[2 * HU + nloc];
  const float bO = bsum[3 * HU + nloc];

  // fragment loaders
  auto loadA = [&](int kb) {
    v8bf lo = *(const v8bf*)(arow + kb);
    v8bf hi = *(const v8bf*)(arow + kb + 8);
    v16bf f;
#pragma unroll
    for (int e = 0; e < 8; ++e) { f[e] = lo[e]; f[8 + e] = hi[e]; }
    return f;
  };
  auto loadB = [&](const bf16_t* w, int kb) {
    return *(const v16bf*)(w + kb);
  };

  v8f creg = {};                        // register-resident cell state

  for (int s = 0; s < T_STEPS; ++s) {
    const int t = T_STEPS - 1 - s;      // sequence is time-reversed
    const int p = s & 1;
    const bf16_t* hin  = hbuf + p * (BATCH * HID);
    bf16_t*       hout = hbuf + (p ^ 1) * (BATCH * HID);

    // ---- stage A tile: h part (32x512 bf16, 16B vector copies) ----
    for (int i = tid; i < BT * (HID / 8); i += THREADS) {
      int m = i >> 6, ch = i & 63;
      *(v8bf*)(As + m * KDIM + ch * 8) =
          *(const v8bf*)(hin + (size_t)(bt * BT + m) * HID + ch * 8);
    }
    // ---- obs part: concat(x*mask, a) -> bf16 ----
    for (int i = tid; i < BT * IND; i += THREADS) {
      int m = i >> 5, j = i & 31;
      int bm = bt * BT + m;
      float v;
      if (j < XD) v = x[((size_t)t * BATCH + bm) * XD + j] *
                      mask[((size_t)t * BATCH + bm) * XD + j];
      else        v = a[((size_t)t * BATCH + bm) * AD + (j - XD)];
      As[m * KDIM + HID + j] = (bf16_t)v;
    }
    __syncthreads();

    // ---- 4-gate WMMA chain, software-pipelined one K-stage deep ----
    v8f accI = {}, accF = {}, accG = {}, accO = {};
    v16bf aC  = loadA(0);
    v16bf biC = loadB(wI, 0), bfC = loadB(wF, 0);
    v16bf bgC = loadB(wG, 0), boC = loadB(wO, 0);
#pragma unroll
    for (int kb = 0; kb < KDIM; kb += 32) {
      v16bf aN = {}, biN = {}, bfN = {}, bgN = {}, boN = {};
      if (kb + 32 < KDIM) {
        aN  = loadA(kb + 32);
        biN = loadB(wI, kb + 32);
        bfN = loadB(wF, kb + 32);
        bgN = loadB(wG, kb + 32);
        boN = loadB(wO, kb + 32);
      }
      accI = __builtin_amdgcn_wmma_f32_16x16x32_bf16(
                 false, aC, false, biC, (short)0, accI, false, false);
      accF = __builtin_amdgcn_wmma_f32_16x16x32_bf16(
                 false, aC, false, bfC, (short)0, accF, false, false);
      accG = __builtin_amdgcn_wmma_f32_16x16x32_bf16(
                 false, aC, false, bgC, (short)0, accG, false, false);
      accO = __builtin_amdgcn_wmma_f32_16x16x32_bf16(
                 false, aC, false, boC, (short)0, accO, false, false);
      aC = aN; biC = biN; bfC = bfN; bgC = bgN; boC = boN;
    }

    // ---- cell update entirely in registers; write h slice to global ----
    // C layout: element r of each acc is (M = r or r+8, N = lane&15); the
    // same (lane,r) indexes the same (batch row, hidden unit) in all 4 accs.
#pragma unroll
    for (int r = 0; r < 8; ++r) {
      float gi = accI[r] + bI;
      float gf = accF[r] + bF;
      float gg = accG[r] + bG;
      float go = accO[r] + bO;
      float cnew = sigmoidf_fast(gf) * creg[r] + sigmoidf_fast(gi) * tanhf(gg);
      creg[r] = cnew;
      float hnew = sigmoidf_fast(go) * tanhf(cnew);
      int Mg = bt * BT + mtile * 16 + ((lane < 16) ? r : r + 8);
      hout[(size_t)Mg * HID + hg * HU + nloc] = (bf16_t)hnew;
    }

    // ---- release h slice, phase-barrier the NH WGs of this batch group ----
    __threadfence();     // make our h writes agent-visible
    __syncthreads();
    if (tid == 0) {
      __hip_atomic_fetch_add(&cnt[bt], 1, __ATOMIC_RELEASE,
                             __HIP_MEMORY_SCOPE_AGENT);
      const int target = NH * (s + 1);
      while (__hip_atomic_load(&cnt[bt], __ATOMIC_ACQUIRE,
                               __HIP_MEMORY_SCOPE_AGENT) < target)
        __builtin_amdgcn_s_sleep(1);
    }
    __syncthreads();
    __threadfence();     // acquire: don't reuse stale h lines next step
  }
}

// ---------------------------------------------------------------------------
// Kernel 2: output heads. mu = exp(h@lin_w^T + lin_b)/10 ; lv = h@lv_w^T+lv_b-5
// Final h lives in buffer 0 (T even). 16 blocks x 8 waves; 1 WMMA tile/wave.
// ---------------------------------------------------------------------------
#define HB 16   // batch rows per heads block

__launch_bounds__(256)
__global__ void heads_kernel(const float* __restrict__ lin_b,
                             const float* __restrict__ lv_b,
                             const char* __restrict__ ws,
                             float* __restrict__ out) {
  const bf16_t* h  = (const bf16_t*)(ws + H_OFF);   // final h, [256][512] bf16
  const bf16_t* HW = (const bf16_t*)(ws + HW_OFF);

  const int bt    = blockIdx.x;
  const int tid   = threadIdx.x;
  const int lane  = tid & 31;
  const int wave  = tid >> 5;
  const int head  = wave >> 2;   // 0 = mu, 1 = log_var
  const int ntile = wave & 3;    // 4 tiles of 16 output cols each

  const bf16_t* W    = HW + (size_t)head * OUTD * HID;
  const float*  bias = head ? lv_b : lin_b;

  const int row   = lane & 15;
  const int k0off = (lane >> 4) * 16;
  const int kh16  = (lane >> 4) * 16;
  const int col   = ntile * 16 + row;

  v8f acc = {};
#pragma unroll
  for (int kb = 0; kb < HID; kb += 32) {
    v16bf afrag;
    v8bf lo = *(const v8bf*)(h + (size_t)(bt * HB + row) * HID + kb + k0off);
    v8bf hi = *(const v8bf*)(h + (size_t)(bt * HB + row) * HID + kb + k0off + 8);
#pragma unroll
    for (int e = 0; e < 8; ++e) { afrag[e] = lo[e]; afrag[8 + e] = hi[e]; }
    v16bf bfrag = *(const v16bf*)(W + (size_t)col * HID + kb + kh16);
    acc = __builtin_amdgcn_wmma_f32_16x16x32_bf16(
              false, afrag, false, bfrag, (short)0, acc, false, false);
  }
#pragma unroll
  for (int r = 0; r < 8; ++r) {
    int M  = (lane < 16) ? r : r + 8;
    int n  = ntile * 16 + (lane & 15);
    int bm = bt * HB + M;
    float v = acc[r] + bias[n];
    if (head == 0) out[(size_t)bm * OUTD + n] = __expf(v) * 0.1f;
    else           out[(size_t)BATCH * OUTD + (size_t)bm * OUTD + n] = v - 5.0f;
  }
}

// ---------------------------------------------------------------------------
// Host entry
// ---------------------------------------------------------------------------
extern "C" void kernel_launch(void* const* d_in, const int* in_sizes, int n_in,
                              void* d_out, int out_size, void* d_ws,
                              size_t ws_size, hipStream_t stream) {
  const float* x     = (const float*)d_in[0];
  const float* a     = (const float*)d_in[1];
  const float* mask  = (const float*)d_in[2];
  const float* W_ih  = (const float*)d_in[3];
  const float* W_hh  = (const float*)d_in[4];
  const float* b_ih  = (const float*)d_in[5];
  const float* b_hh  = (const float*)d_in[6];
  const float* lin_w = (const float*)d_in[7];
  const float* lin_b = (const float*)d_in[8];
  const float* lv_w  = (const float*)d_in[9];
  const float* lv_b  = (const float*)d_in[10];
  char*  ws  = (char*)d_ws;
  float* out = (float*)d_out;

  prep_kernel<<<512, 256, 0, stream>>>(W_ih, W_hh, lin_w, lv_w, ws);
  lstm_kernel<<<NB * NH, THREADS, 0, stream>>>(x, a, mask, b_ih, b_hh, ws);
  heads_kernel<<<BATCH / HB, 256, 0, stream>>>(lin_b, lv_b, ws, out);
}
